// Wav2Vec2BertSelfAttention_62088047231258
// MI455X (gfx1250) — matrix operand order repaired and checked by reading, not compiled, not run on hardware
//
#include <hip/hip_runtime.h>
#include <hip/hip_bf16.h>
#include <math.h>

// ---------------------------------------------------------------------------
// Wav2Vec2Bert self-attention, relative_key position bias.
// B=4, S=1024, HID=1024, NH=16, HD=64, LEFT=64, RIGHT=8, NUM_POS=73 (pad 80).
// bf16 WMMA (v_wmma_f32_16x16x32_bf16) everywhere, f32 accumulate.
// 32x64 tiles per wave; A-stream double-buffered, B loaded per step as one
// clause; pointer-increment addressing to keep VGPR pressure below spill.
// ---------------------------------------------------------------------------

#define B_    4
#define S_    1024
#define HID_  1024
#define NH_   16
#define HD_   64
#define LEFT_ 64
#define RIGHT_ 8
#define NPOS_ 73
#define NPAD_ 80
#define BH_   (B_ * NH_)

typedef __bf16 bf16;
typedef __attribute__((ext_vector_type(16))) __bf16 bf16x16;
typedef __attribute__((ext_vector_type(8)))  __bf16 bf16x8;
typedef __attribute__((ext_vector_type(8)))  float  v8f;
typedef __attribute__((ext_vector_type(8)))  float  f32x8;

#define WMMA_BF16(a, b, c) \
    __builtin_amdgcn_wmma_f32_16x16x32_bf16(false, (a), false, (b), (short)0, (c), false, false)

// ----------------------------- fragment loads ------------------------------
// A-matrix 16x32 bf16 layout (ISA 7.12.2): lane = hf*16 + m,
// element i: K = 16*(i/8) + 8*hf + (i%8).
// Per-lane pointer p = &A[m0 + lane%16][k0] + 8*hf ; chunks at p and p+16.
static __device__ __forceinline__ bf16x16 frag_a(const bf16* __restrict__ p) {
    bf16x8 lo = *(const bf16x8*)(p);
    bf16x8 hi = *(const bf16x8*)(p + 16);
    bf16x16 r;
#pragma unroll
    for (int i = 0; i < 8; ++i) { r[i] = lo[i]; r[8 + i] = hi[i]; }
    return r;
}

// Same A fragment, f32 source (probs), converted on the fly.
static __device__ __forceinline__ bf16x16 frag_a_f32(const float* __restrict__ p) {
    f32x8 lo = *(const f32x8*)(p);
    f32x8 hi = *(const f32x8*)(p + 16);
    bf16x16 r;
#pragma unroll
    for (int i = 0; i < 8; ++i) { r[i] = (bf16)lo[i]; r[8 + i] = (bf16)hi[i]; }
    return r;
}

// B-matrix 32x16 bf16 layout: lane = hf*16 + n, element i: K = 16*(i/8) + 8*hf + (i%8).
// Per-lane pointer q = &Wrow[n0 + lane%16][k0] + 8*hf ; chunks at q and q+16.
static __device__ __forceinline__ bf16x16 frag_b(const bf16* __restrict__ p) {
    bf16x8 lo = *(const bf16x8*)(p);
    bf16x8 hi = *(const bf16x8*)(p + 16);
    bf16x16 r;
#pragma unroll
    for (int i = 0; i < 8; ++i) { r[i] = lo[i]; r[8 + i] = hi[i]; }
    return r;
}

// ------------------------------ conversions --------------------------------
__global__ void cvt_f32_bf16(const float* __restrict__ src, bf16* __restrict__ dst, int n) {
    for (int i = blockIdx.x * blockDim.x + threadIdx.x; i < n; i += gridDim.x * blockDim.x)
        dst[i] = (bf16)src[i];
}

__global__ void pad_dist_emb(const float* __restrict__ de, bf16* __restrict__ dst) {
    int i = blockIdx.x * blockDim.x + threadIdx.x;
    if (i >= NPAD_ * HD_) return;
    int p = i >> 6, d = i & 63;
    dst[i] = (p < NPOS_) ? (bf16)de[p * HD_ + d] : (bf16)0.0f;
}

// ----------------------- generic GEMM:  C = A @ W^T + b --------------------
// A: [M,K] bf16 row-major.  W: [N,K] bf16 row-major.  bias: [N] f32.
// Wave tile: 32 (M) x 64 (N).  A-stream double-buffered over K (32/step).
// mode 0: f32 out row-major [M,N]
// mode 1: bf16 out scattered to [b,h,s,d]   (Q16 / K16)
// mode 2: bf16 out scattered to [b,h,d,s]   (V transposed)
__global__ __launch_bounds__(128, 1) void
gemm_xwT(const bf16* __restrict__ A, const bf16* __restrict__ W,
         const float* __restrict__ bias,
         float* __restrict__ outF, bf16* __restrict__ outB,
         int M, int N, int K, int mode) {
    const int lane = threadIdx.x & 31;
    const int wid  = blockIdx.x * (blockDim.x >> 5) + (threadIdx.x >> 5);
    const int n4   = N >> 6;
    const int mb   = (wid / n4) << 5;     // 32 rows
    const int nb   = (wid % n4) << 6;     // 64 cols
    if (mb >= M) return;

    const int ml = lane & 15, hf = lane >> 4;
    const bf16* pa0 = A + (size_t)(mb + ml) * K + 8 * hf;
    const bf16* pa1 = A + (size_t)(mb + 16 + ml) * K + 8 * hf;
    const bf16* pb0 = W + (size_t)(nb + ml) * K + 8 * hf;
    const bf16* pb1 = W + (size_t)(nb + 16 + ml) * K + 8 * hf;
    const bf16* pb2 = W + (size_t)(nb + 32 + ml) * K + 8 * hf;
    const bf16* pb3 = W + (size_t)(nb + 48 + ml) * K + 8 * hf;

    v8f acc[2][4] = {};
    bf16x16 a0 = frag_a(pa0);
    bf16x16 a1 = frag_a(pa1);

    const int steps = K >> 5;
    for (int s = 0; s < steps - 1; ++s) {
        // B fragments for current k (one clause of 8 b128 loads)
        bf16x16 b0 = frag_b(pb0);
        bf16x16 b1 = frag_b(pb1);
        bf16x16 b2 = frag_b(pb2);
        bf16x16 b3 = frag_b(pb3);
        pb0 += 32; pb1 += 32; pb2 += 32; pb3 += 32;
        // next-step A fragments: in flight across the WMMA group
        pa0 += 32; pa1 += 32;
        __builtin_prefetch(pa0 + 32, 0, 3);
        bf16x16 na0 = frag_a(pa0);
        bf16x16 na1 = frag_a(pa1);

        acc[0][0] = WMMA_BF16(a0, b0, acc[0][0]);
        acc[1][0] = WMMA_BF16(a1, b0, acc[1][0]);
        acc[0][1] = WMMA_BF16(a0, b1, acc[0][1]);
        acc[1][1] = WMMA_BF16(a1, b1, acc[1][1]);
        acc[0][2] = WMMA_BF16(a0, b2, acc[0][2]);
        acc[1][2] = WMMA_BF16(a1, b2, acc[1][2]);
        acc[0][3] = WMMA_BF16(a0, b3, acc[0][3]);
        acc[1][3] = WMMA_BF16(a1, b3, acc[1][3]);

        a0 = na0; a1 = na1;
    }
    {   // last K step
        bf16x16 b0 = frag_b(pb0);
        bf16x16 b1 = frag_b(pb1);
        bf16x16 b2 = frag_b(pb2);
        bf16x16 b3 = frag_b(pb3);
        acc[0][0] = WMMA_BF16(a0, b0, acc[0][0]);
        acc[1][0] = WMMA_BF16(a1, b0, acc[1][0]);
        acc[0][1] = WMMA_BF16(a0, b1, acc[0][1]);
        acc[1][1] = WMMA_BF16(a1, b1, acc[1][1]);
        acc[0][2] = WMMA_BF16(a0, b2, acc[0][2]);
        acc[1][2] = WMMA_BF16(a1, b2, acc[1][2]);
        acc[0][3] = WMMA_BF16(a0, b3, acc[0][3]);
        acc[1][3] = WMMA_BF16(a1, b3, acc[1][3]);
    }

#pragma unroll
    for (int mi = 0; mi < 2; ++mi)
#pragma unroll
        for (int t = 0; t < 4; ++t)
#pragma unroll
            for (int j = 0; j < 8; ++j) {
                const int m = mb + 16 * mi + j + 8 * hf;
                const int n = nb + 16 * t + ml;
                const float v = acc[mi][t][j] + bias[n];
                if (mode == 0) {
                    outF[(size_t)m * N + n] = v;
                } else {
                    const int bb = m >> 10, ss = m & 1023, hh = n >> 6, dd = n & 63;
                    if (mode == 1)
                        outB[((((size_t)bb * NH_ + hh) * S_) + ss) * HD_ + dd] = (bf16)v;
                    else
                        outB[((((size_t)bb * NH_ + hh) * HD_) + dd) * S_ + ss] = (bf16)v;
                }
            }
}

// -------------------- rel bias:  R[row,p] = Q[row,:]·DE[p,:] ---------------
__global__ __launch_bounds__(128, 1) void
rel_gemm(const bf16* __restrict__ Q, const bf16* __restrict__ DE,
         float* __restrict__ R) {
    const int lane = threadIdx.x & 31;
    const int wid  = blockIdx.x * (blockDim.x >> 5) + (threadIdx.x >> 5);
    const int mb   = wid << 4;
    if (mb >= BH_ * S_) return;

    const int ml = lane & 15, hf = lane >> 4;
    v8f acc[5] = {};
#pragma unroll
    for (int k = 0; k < HD_; k += 32) {
        bf16x16 a = frag_a(Q + (size_t)(mb + ml) * HD_ + k + 8 * hf);
#pragma unroll
        for (int t = 0; t < 5; ++t) {
            bf16x16 b = frag_b(DE + (size_t)(16 * t + ml) * HD_ + k + 8 * hf);
            acc[t] = WMMA_BF16(a, b, acc[t]);
        }
    }
#pragma unroll
    for (int t = 0; t < 5; ++t)
#pragma unroll
        for (int j = 0; j < 8; ++j)
            R[(size_t)(mb + j + 8 * hf) * NPAD_ + 16 * t + ml] = acc[t][j];
}

// --------- scores = (Q K^T + gather(rel)) * 1/sqrt(HD) -> probs buf --------
// Wave tile: 32 (l) x 64 (r); K-loop (HD=64) fully unrolled, B shared by rows.
__global__ __launch_bounds__(128, 1) void
scores_gemm(const bf16* __restrict__ Q, const bf16* __restrict__ Kt,
            const float* __restrict__ R, float* __restrict__ scr) {
    const int lane = threadIdx.x & 31;
    const int wid  = blockIdx.x * (blockDim.x >> 5) + (threadIdx.x >> 5);
    const int rs = S_ >> 6, ls = S_ >> 5;
    const int bh  = wid / (rs * ls);
    if (bh >= BH_) return;
    const int rem = wid % (rs * ls);
    const int lb  = (rem / rs) << 5;
    const int rb  = (rem % rs) << 6;

    const int ml = lane & 15, hf = lane >> 4;
    const bf16* Qb = Q  + (size_t)bh * S_ * HD_;
    const bf16* Kb = Kt + (size_t)bh * S_ * HD_;

    v8f acc[2][4] = {};
#pragma unroll
    for (int k = 0; k < HD_; k += 32) {
        bf16x16 a0 = frag_a(Qb + (size_t)(lb + ml) * HD_ + k + 8 * hf);
        bf16x16 a1 = frag_a(Qb + (size_t)(lb + 16 + ml) * HD_ + k + 8 * hf);
#pragma unroll
        for (int t = 0; t < 4; ++t) {
            bf16x16 b = frag_b(Kb + (size_t)(rb + 16 * t + ml) * HD_ + k + 8 * hf);
            acc[0][t] = WMMA_BF16(a0, b, acc[0][t]);
            acc[1][t] = WMMA_BF16(a1, b, acc[1][t]);
        }
    }

    const float inv = 0.125f;  // 1/sqrt(64)
#pragma unroll
    for (int mi = 0; mi < 2; ++mi)
#pragma unroll
        for (int t = 0; t < 4; ++t)
#pragma unroll
            for (int j = 0; j < 8; ++j) {
                const int l = lb + 16 * mi + j + 8 * hf;
                const int r = rb + 16 * t + ml;
                int dist = r - l;
                dist = dist < -LEFT_ ? -LEFT_ : (dist > RIGHT_ ? RIGHT_ : dist);
                const float rel = R[((size_t)bh * S_ + l) * NPAD_ + dist + LEFT_];
                scr[(((size_t)bh * S_) + l) * S_ + r] = (acc[mi][t][j] + rel) * inv;
            }
}

// ------------------------- softmax rows, in place --------------------------
__global__ void softmax_rows(float* __restrict__ P) {
    __shared__ float red[256];
    float* p = P + (size_t)blockIdx.x * S_;
    const int tid = threadIdx.x;

    float x[4], mx = -INFINITY;
#pragma unroll
    for (int i = 0; i < 4; ++i) { x[i] = p[tid + i * 256]; mx = fmaxf(mx, x[i]); }
    red[tid] = mx; __syncthreads();
    for (int s = 128; s > 0; s >>= 1) {
        if (tid < s) red[tid] = fmaxf(red[tid], red[tid + s]);
        __syncthreads();
    }
    mx = red[0]; __syncthreads();

    float sum = 0.0f;
#pragma unroll
    for (int i = 0; i < 4; ++i) { x[i] = __expf(x[i] - mx); sum += x[i]; }
    red[tid] = sum; __syncthreads();
    for (int s = 128; s > 0; s >>= 1) {
        if (tid < s) red[tid] += red[tid + s];
        __syncthreads();
    }
    const float inv = 1.0f / red[0];
#pragma unroll
    for (int i = 0; i < 4; ++i) p[tid + i * 256] = x[i] * inv;
}

// ------------- ctx = probs @ V  (A from f32, B from V^T bf16) --------------
// Wave tile: 32 (l) x 64 (d = full head). A-stream double-buffered over r.
__global__ __launch_bounds__(128, 1) void
ctx_gemm(const float* __restrict__ P, const bf16* __restrict__ VT,
         bf16* __restrict__ CTX) {
    const int lane = threadIdx.x & 31;
    const int wid  = blockIdx.x * (blockDim.x >> 5) + (threadIdx.x >> 5);
    const int lt = S_ >> 5;
    const int bh = wid / lt;
    if (bh >= BH_) return;
    const int lb = (wid % lt) << 5;

    const int ml = lane & 15, hf = lane >> 4;
    const float* Pb = P  + (size_t)bh * S_ * S_;
    const bf16*  Vb = VT + (size_t)bh * HD_ * S_;

    const float* pa0 = Pb + (size_t)(lb + ml) * S_ + 8 * hf;
    const float* pa1 = Pb + (size_t)(lb + 16 + ml) * S_ + 8 * hf;
    const bf16*  pb0 = Vb + (size_t)(ml) * S_ + 8 * hf;
    const bf16*  pb1 = Vb + (size_t)(16 + ml) * S_ + 8 * hf;
    const bf16*  pb2 = Vb + (size_t)(32 + ml) * S_ + 8 * hf;
    const bf16*  pb3 = Vb + (size_t)(48 + ml) * S_ + 8 * hf;

    v8f acc[2][4] = {};
    bf16x16 a0 = frag_a_f32(pa0);
    bf16x16 a1 = frag_a_f32(pa1);

    const int steps = S_ >> 5;
    for (int s = 0; s < steps - 1; ++s) {
        bf16x16 b0 = frag_b(pb0);
        bf16x16 b1 = frag_b(pb1);
        bf16x16 b2 = frag_b(pb2);
        bf16x16 b3 = frag_b(pb3);
        pb0 += 32; pb1 += 32; pb2 += 32; pb3 += 32;
        pa0 += 32; pa1 += 32;
        __builtin_prefetch(pa0 + 32, 0, 3);
        bf16x16 na0 = frag_a_f32(pa0);
        bf16x16 na1 = frag_a_f32(pa1);

        acc[0][0] = WMMA_BF16(a0, b0, acc[0][0]);
        acc[1][0] = WMMA_BF16(a1, b0, acc[1][0]);
        acc[0][1] = WMMA_BF16(a0, b1, acc[0][1]);
        acc[1][1] = WMMA_BF16(a1, b1, acc[1][1]);
        acc[0][2] = WMMA_BF16(a0, b2, acc[0][2]);
        acc[1][2] = WMMA_BF16(a1, b2, acc[1][2]);
        acc[0][3] = WMMA_BF16(a0, b3, acc[0][3]);
        acc[1][3] = WMMA_BF16(a1, b3, acc[1][3]);

        a0 = na0; a1 = na1;
    }
    {
        bf16x16 b0 = frag_b(pb0);
        bf16x16 b1 = frag_b(pb1);
        bf16x16 b2 = frag_b(pb2);
        bf16x16 b3 = frag_b(pb3);
        acc[0][0] = WMMA_BF16(a0, b0, acc[0][0]);
        acc[1][0] = WMMA_BF16(a1, b0, acc[1][0]);
        acc[0][1] = WMMA_BF16(a0, b1, acc[0][1]);
        acc[1][1] = WMMA_BF16(a1, b1, acc[1][1]);
        acc[0][2] = WMMA_BF16(a0, b2, acc[0][2]);
        acc[1][2] = WMMA_BF16(a1, b2, acc[1][2]);
        acc[0][3] = WMMA_BF16(a0, b3, acc[0][3]);
        acc[1][3] = WMMA_BF16(a1, b3, acc[1][3]);
    }

    const int bb = bh / NH_, hh = bh % NH_;
#pragma unroll
    for (int mi = 0; mi < 2; ++mi)
#pragma unroll
        for (int t = 0; t < 4; ++t)
#pragma unroll
            for (int j = 0; j < 8; ++j) {
                const int l = lb + 16 * mi + j + 8 * hf;
                const int d = 16 * t + ml;
                CTX[(((size_t)bb * S_ + l) * HID_) + hh * HD_ + d] = (bf16)acc[mi][t][j];
            }
}

// --------------------------------- launch ----------------------------------
extern "C" void kernel_launch(void* const* d_in, const int* in_sizes, int n_in,
                              void* d_out, int out_size, void* d_ws, size_t ws_size,
                              hipStream_t stream) {
    const float* hs = (const float*)d_in[0];
    const float* Wq = (const float*)d_in[1];
    const float* bq = (const float*)d_in[2];
    const float* Wk = (const float*)d_in[3];
    const float* bk = (const float*)d_in[4];
    const float* Wv = (const float*)d_in[5];
    const float* bv = (const float*)d_in[6];
    const float* Wo = (const float*)d_in[7];
    const float* bo = (const float*)d_in[8];
    const float* de = (const float*)d_in[9];

    float* out   = (float*)d_out;
    float* probs = out + (size_t)B_ * S_ * HID_;   // scores -> probs, in place

    char* ws = (char*)d_ws;
    size_t off = 0;
    auto alloc = [&](size_t bytes) -> void* {
        void* p = ws + off;
        off += (bytes + 255) & ~(size_t)255;
        return p;
    };
    const size_t MROWS = (size_t)B_ * S_;              // 4096
    bf16* X16   = (bf16*)alloc(MROWS * HID_ * 2);
    bf16* Wq16  = (bf16*)alloc((size_t)HID_ * HID_ * 2);
    bf16* Wk16  = (bf16*)alloc((size_t)HID_ * HID_ * 2);
    bf16* Wv16  = (bf16*)alloc((size_t)HID_ * HID_ * 2);
    bf16* Wo16  = (bf16*)alloc((size_t)HID_ * HID_ * 2);
    bf16* Q16   = (bf16*)alloc((size_t)BH_ * S_ * HD_ * 2);
    bf16* K16   = (bf16*)alloc((size_t)BH_ * S_ * HD_ * 2);
    bf16* VT16  = (bf16*)alloc((size_t)BH_ * HD_ * S_ * 2);
    bf16* CTX16 = (bf16*)alloc(MROWS * HID_ * 2);
    float* R    = (float*)alloc((size_t)BH_ * S_ * NPAD_ * 4);
    bf16* DE16  = (bf16*)alloc((size_t)NPAD_ * HD_ * 2);

    // 1) bf16 conversions
    cvt_f32_bf16<<<2048, 256, 0, stream>>>(hs, X16, (int)(MROWS * HID_));
    cvt_f32_bf16<<<1024, 256, 0, stream>>>(Wq, Wq16, HID_ * HID_);
    cvt_f32_bf16<<<1024, 256, 0, stream>>>(Wk, Wk16, HID_ * HID_);
    cvt_f32_bf16<<<1024, 256, 0, stream>>>(Wv, Wv16, HID_ * HID_);
    cvt_f32_bf16<<<1024, 256, 0, stream>>>(Wo, Wo16, HID_ * HID_);
    pad_dist_emb<<<(NPAD_ * HD_ + 255) / 256, 256, 0, stream>>>(de, DE16);

    // 2) Q/K/V projections: waves = (4096/32)*(1024/64) = 2048 -> 512 blocks
    gemm_xwT<<<512, 128, 0, stream>>>(X16, Wq16, bq, nullptr, Q16, (int)MROWS, HID_, HID_, 1);
    gemm_xwT<<<512, 128, 0, stream>>>(X16, Wk16, bk, nullptr, K16, (int)MROWS, HID_, HID_, 1);
    gemm_xwT<<<512, 128, 0, stream>>>(X16, Wv16, bv, nullptr, VT16, (int)MROWS, HID_, HID_, 2);

    // 3) relative-position bias table R[row, 80]: waves = 4096 -> 1024 blocks
    rel_gemm<<<1024, 128, 0, stream>>>(Q16, DE16, R);

    // 4) scores (+rel, *1/8) -> probs region. waves = 64*32*16 = 32768
    scores_gemm<<<8192, 128, 0, stream>>>(Q16, K16, R, probs);

    // 5) softmax over r, in place (one block per (b,h,l) row)
    softmax_rows<<<BH_ * S_, 256, 0, stream>>>(probs);

    // 6) ctx = probs @ V. waves = 64*32 = 2048 -> 512 blocks
    ctx_gemm<<<512, 128, 0, stream>>>(probs, VT16, CTX16);

    // 7) output projection -> f32 out
    gemm_xwT<<<512, 128, 0, stream>>>(CTX16, Wo16, bo, out, nullptr, (int)MROWS, HID_, HID_, 0);
}
